// SemanticMapSelfLSTMModel_24206435680311
// MI455X (gfx1250) — matrix operand order repaired and checked by reading, not compiled, not run on hardware
//
#include <hip/hip_runtime.h>
#include <hip/hip_bf16.h>
#include <stdint.h>

typedef __attribute__((ext_vector_type(16))) __bf16 v16bf;
typedef __attribute__((ext_vector_type(8)))  __bf16 v8bf;
typedef __attribute__((ext_vector_type(8)))  float  v8f;

#define N_ROWS 8192
#define L_OBS  20
#define P_PRED 30
#define E_DIM  64
#define H_DIM  128
#define G_DIM  512      // 4*H
#define WAVES  4        // 4 waves * 16 batch rows = 64 rows per block
#define HT_STRIDE 144   // padded bf16 row stride for h-tile (32B aligned rows)

union BF2 { __bf16 b[2]; uint32_t u; };

__device__ __forceinline__ float sigm_(float x) { return 1.0f / (1.0f + __expf(-x)); }
__device__ __forceinline__ float tanh_(float x) {
  float t = __expf(-2.0f * fabsf(x));
  float r = (1.0f - t) / (1.0f + t);
  return copysignf(r, x);
}

// k-offset within a 32-wide K block for 16-bit WMMA A fragments:
// lanes 0-15 hold K = {0..15}, lanes 16-31 hold K = {16..31}
__device__ __forceinline__ int amap_(int hh, int i) {
  return hh * 16 + i;
}

// ---------------------------------------------------------------------------
// Kernel 0: stage W_hh^T as bf16 WMMA A-fragments (fragment-linear) in ws.
// 32 gate tiles * 4 kfrags * 32 lanes * 16 elems = 65536 bf16 (128 KB).
// ---------------------------------------------------------------------------
__global__ void stage_whhT_kernel(const float* __restrict__ Whh,
                                  __bf16* __restrict__ dst) {
  int idx = blockIdx.x * blockDim.x + threadIdx.x;
  if (idx >= 32 * 4 * 32 * 16) return;
  int i    = idx & 15;
  int lane = (idx >> 4) & 31;
  int kf   = (idx >> 9) & 3;
  int tg   = idx >> 11;
  int k    = 32 * kf + amap_(lane >> 4, i);
  int gcol = 16 * tg + (lane & 15);
  dst[idx] = (__bf16)Whh[k * G_DIM + gcol];
}

// ---------------------------------------------------------------------------
// Kernel 1: img_proj[n][j] = img[n] . W_pred[H:][:,j] + b_pred[j]
// Memory-bound (67 MB stream). One wave per row, wave32 shuffle reduction.
// ---------------------------------------------------------------------------
__global__ void img_proj_kernel(const float* __restrict__ img,
                                const float* __restrict__ Wpred,
                                const float* __restrict__ bpred,
                                float* __restrict__ out) {
  int wave = threadIdx.x >> 5, lane = threadIdx.x & 31;
  int row  = blockIdx.x * (blockDim.x >> 5) + wave;
  const float4* ip = (const float4*)(img + (size_t)row * 2048);
  float a0 = 0.0f, a1 = 0.0f;
#pragma unroll
  for (int j = 0; j < 16; ++j) {
    int q = j * 32 + lane;                  // float4 index 0..511
    float4 v = ip[q];
    const float2* wr = (const float2*)(Wpred + (size_t)(H_DIM + q * 4) * 2);
    float2 w0 = wr[0], w1 = wr[1], w2 = wr[2], w3 = wr[3];
    a0 += v.x * w0.x + v.y * w1.x + v.z * w2.x + v.w * w3.x;
    a1 += v.x * w0.y + v.y * w1.y + v.z * w2.y + v.w * w3.y;
  }
#pragma unroll
  for (int off = 16; off > 0; off >>= 1) {
    a0 += __shfl_down(a0, off, 32);
    a1 += __shfl_down(a1, off, 32);
  }
  if (lane == 0) {
    out[row * 2 + 0] = a0 + bpred[0];
    out[row * 2 + 1] = a1 + bpred[1];
  }
}

// ---------------------------------------------------------------------------
// Kernel 2: full LSTM rollout. Each wave owns 16 batch rows. c lives in f32
// registers (WMMA C/D layout, batch across lanes); h lives in a per-wave bf16
// LDS transpose tile that feeds the next step's B-fragments directly.
// ---------------------------------------------------------------------------
__global__ void __launch_bounds__(WAVES * 32) lstm_kernel(
    const float* __restrict__ obs_pos, const float* __restrict__ obs_rel,
    const int* __restrict__ hist_sz, const float* __restrict__ h0,
    const float* __restrict__ Wemb, const float* __restrict__ bemb,
    const float* __restrict__ Wih, const float* __restrict__ bih,
    const float* __restrict__ bhh, const float* __restrict__ Wpred,
    const float* __restrict__ img_proj, const __bf16* __restrict__ whhT,
    float* __restrict__ out) {
  // W_ih^T fragments: 32 gate tiles * 2 kfrags * 512 bf16 = 64 KB
  __shared__ __attribute__((aligned(32))) __bf16 s_wih[32 * 2 * 512];
  // per-wave h transpose tile: 16 batch rows x 128 H cols (padded)
  __shared__ __attribute__((aligned(32))) __bf16 s_ht[WAVES][16 * HT_STRIDE];
  __shared__ float s_bias[G_DIM];
  __shared__ float s_we0[E_DIM], s_we1[E_DIM], s_web[E_DIM];
  __shared__ float s_wp[H_DIM * 2];
  __shared__ float s_h0[H_DIM];

  const int tid = threadIdx.x;

  // ---- stage weights into LDS (one-time, cooperative) ----
  for (int idx = tid; idx < 32 * 2 * 512; idx += WAVES * 32) {
    int i = idx & 15, lane = (idx >> 4) & 31, kf = (idx >> 9) & 1, tg = idx >> 10;
    int k = 32 * kf + amap_(lane >> 4, i);
    s_wih[idx] = (__bf16)Wih[k * G_DIM + 16 * tg + (lane & 15)];
  }
  for (int g = tid; g < G_DIM; g += WAVES * 32) s_bias[g] = bih[g] + bhh[g];
  for (int k = tid; k < E_DIM; k += WAVES * 32) {
    s_we0[k] = Wemb[k]; s_we1[k] = Wemb[E_DIM + k]; s_web[k] = bemb[k];
  }
  for (int q = tid; q < H_DIM * 2; q += WAVES * 32) s_wp[q] = Wpred[q];
  for (int q = tid; q < H_DIM; q += WAVES * 32) s_h0[q] = h0[q];
  __syncthreads();

  const int wave = tid >> 5, lane = tid & 31;
  const int n = lane & 15;        // batch column within tile
  const int hh = lane >> 4;       // lane half
  const long brow = ((long)blockIdx.x * WAVES + wave) * 16 + n;
  const int hist = hist_sz[brow];
  __bf16* myt = &s_ht[wave][0];

  // ---- init state: c (f32 regs) = h (LDS tile, bf16) = h0 ----
  float c[8][8];
#pragma unroll
  for (int t = 0; t < 8; ++t) {
#pragma unroll
    for (int r = 0; r < 8; r += 2) {
      float v0 = s_h0[16 * t + 8 * hh + r];
      float v1 = s_h0[16 * t + 8 * hh + r + 1];
      c[t][r] = v0; c[t][r + 1] = v1;
      BF2 p2; p2.b[0] = (__bf16)v0; p2.b[1] = (__bf16)v1;
      *(uint32_t*)&myt[n * HT_STRIDE + 16 * t + 8 * hh + r] = p2.u;
    }
  }

  // ---- one LSTM step: gates^T = W_ih^T x e^T + W_hh^T x h^T + bias ----
  auto lstm_step = [&](float rel0, float rel1, bool upd) {
    // Opaque memory barrier: stops LICM from hoisting the (loop-invariant)
    // weight-fragment loads out of the step loop and spilling them to scratch.
    asm volatile("" ::: "memory");

    // e^T B-fragments computed directly in registers (lane = its batch row)
    v16bf ef[2];
#pragma unroll
    for (int kf = 0; kf < 2; ++kf) {
      int kb = 32 * kf + 16 * hh;
#pragma unroll
      for (int ii = 0; ii < 16; ++ii) {
        int k = kb + ii;
        float ev = fmaf(rel0, s_we0[k], fmaf(rel1, s_we1[k], s_web[k]));
        ef[kf][ii] = (__bf16)fmaxf(ev, 0.0f);
      }
    }
    // h^T B-fragments from per-wave LDS tile (contiguous 32B per lane)
    v16bf hf[4];
#pragma unroll
    for (int kf = 0; kf < 4; ++kf)
      hf[kf] = *(const v16bf*)&myt[n * HT_STRIDE + 32 * kf + 16 * hh];

#pragma unroll
    for (int t = 0; t < 8; ++t) {
      v8f acc[4];
#pragma unroll
      for (int gi = 0; gi < 4; ++gi) {
        int tg = t + 8 * gi;
        const float4* bp = (const float4*)&s_bias[16 * tg + 8 * hh];
        float4 b0 = bp[0], b1 = bp[1];
        v8f a;
        a[0] = b0.x; a[1] = b0.y; a[2] = b0.z; a[3] = b0.w;
        a[4] = b1.x; a[5] = b1.y; a[6] = b1.z; a[7] = b1.w;
        // batch the 4 W_hh fragment loads (L2) so they clause together
        v16bf wb[4];
#pragma unroll
        for (int kf = 0; kf < 4; ++kf)
          wb[kf] = *(const v16bf*)&whhT[((tg * 4 + kf) * 32 + lane) * 16];
#pragma unroll
        for (int kf = 0; kf < 4; ++kf)
          a = __builtin_amdgcn_wmma_f32_16x16x32_bf16(
              false, wb[kf], false, hf[kf], (short)0, a, false, false);
#pragma unroll
        for (int kf = 0; kf < 2; ++kf) {
          v16bf wa = *(const v16bf*)&s_wih[((tg * 2 + kf) * 32 + lane) * 16];
          a = __builtin_amdgcn_wmma_f32_16x16x32_bf16(
              false, wa, false, ef[kf], (short)0, a, false, false);
        }
        acc[gi] = a;
      }
      // gate order: i, f, g, o
#pragma unroll
      for (int r = 0; r < 8; r += 2) {
        float cn0 = sigm_(acc[1][r]) * c[t][r] + sigm_(acc[0][r]) * tanh_(acc[2][r]);
        float cn1 = sigm_(acc[1][r + 1]) * c[t][r + 1] +
                    sigm_(acc[0][r + 1]) * tanh_(acc[2][r + 1]);
        float hn0 = sigm_(acc[3][r]) * tanh_(cn0);
        float hn1 = sigm_(acc[3][r + 1]) * tanh_(cn1);
        c[t][r]     = upd ? cn0 : c[t][r];      // lane-uniform select
        c[t][r + 1] = upd ? cn1 : c[t][r + 1];
        BF2 p2; p2.b[0] = (__bf16)hn0; p2.b[1] = (__bf16)hn1;
        if (upd)  // masked h update = skip the tile store (no WMMA in region)
          *(uint32_t*)&myt[n * HT_STRIDE + 16 * t + 8 * hh + r] = p2.u;
      }
    }
  };

  // ---- observation phase: 19 masked steps ----
  const float2* relp = (const float2*)obs_rel;
#pragma clang loop unroll(disable)
  for (int s = 0; s < L_OBS - 1; ++s) {
    float2 rv = relp[brow * L_OBS + (s + 1)];
    lstm_step(rv.x, rv.y, hist > (L_OBS - 1 - s));
  }

  // ---- prediction phase: 30 steps ----
  const float ip0 = img_proj[brow * 2 + 0];
  const float ip1 = img_proj[brow * 2 + 1];
  const float2* posp = (const float2*)obs_pos;
  float2 pv = posp[brow * L_OBS + (L_OBS - 1)];
  float p0 = pv.x, p1 = pv.y;

#pragma clang loop unroll(disable)
  for (int p = 0; p < P_PRED; ++p) {
    // rel = h @ Wp_h + img_proj: each half-lane sums its 64 H cols (read back
    // from the bf16 tile), then combines with the other half via xor-16.
    float a0 = 0.0f, a1 = 0.0f;
#pragma unroll
    for (int t = 0; t < 8; ++t) {
      v8bf hv = *(const v8bf*)&myt[n * HT_STRIDE + 16 * t + 8 * hh];
      const float4* wp = (const float4*)&s_wp[(16 * t + 8 * hh) * 2];
      float4 w0 = wp[0], w1 = wp[1], w2 = wp[2], w3 = wp[3];
      a0 = fmaf((float)hv[0], w0.x, a0); a1 = fmaf((float)hv[0], w0.y, a1);
      a0 = fmaf((float)hv[1], w0.z, a0); a1 = fmaf((float)hv[1], w0.w, a1);
      a0 = fmaf((float)hv[2], w1.x, a0); a1 = fmaf((float)hv[2], w1.y, a1);
      a0 = fmaf((float)hv[3], w1.z, a0); a1 = fmaf((float)hv[3], w1.w, a1);
      a0 = fmaf((float)hv[4], w2.x, a0); a1 = fmaf((float)hv[4], w2.y, a1);
      a0 = fmaf((float)hv[5], w2.z, a0); a1 = fmaf((float)hv[5], w2.w, a1);
      a0 = fmaf((float)hv[6], w3.x, a0); a1 = fmaf((float)hv[6], w3.y, a1);
      a0 = fmaf((float)hv[7], w3.z, a0); a1 = fmaf((float)hv[7], w3.w, a1);
    }
    a0 += __shfl_xor(a0, 16, 32);
    a1 += __shfl_xor(a1, 16, 32);
    float r0 = a0 + ip0, r1 = a1 + ip1;
    p0 += r0; p1 += r1;
    if (hh == 0) {
      float2 o2 = make_float2(p0, p1);
      *(float2*)&out[(brow * P_PRED + p) * 2] = o2;
    }
    lstm_step(r0, r1, true);
  }
}

// ---------------------------------------------------------------------------
extern "C" void kernel_launch(void* const* d_in, const int* in_sizes, int n_in,
                              void* d_out, int out_size, void* d_ws, size_t ws_size,
                              hipStream_t stream) {
  const float* img   = (const float*)d_in[0];
  const float* opos  = (const float*)d_in[1];
  const float* orel  = (const float*)d_in[2];
  const int*   ohs   = (const int*)d_in[3];
  const float* h0    = (const float*)d_in[4];
  const float* Wemb  = (const float*)d_in[5];
  const float* bemb  = (const float*)d_in[6];
  const float* Wih   = (const float*)d_in[7];
  const float* Whh   = (const float*)d_in[8];
  const float* bih   = (const float*)d_in[9];
  const float* bhh   = (const float*)d_in[10];
  const float* Wpred = (const float*)d_in[11];
  const float* bpred = (const float*)d_in[12];
  float* out = (float*)d_out;

  float*  ws_img = (float*)d_ws;                                       // N*2 f32
  __bf16* ws_whh = (__bf16*)((char*)d_ws + N_ROWS * 2 * sizeof(float)); // 128 KB

  stage_whhT_kernel<<<256, 256, 0, stream>>>(Whh, ws_whh);
  img_proj_kernel<<<N_ROWS / 8, 256, 0, stream>>>(img, Wpred, bpred, ws_img);
  lstm_kernel<<<N_ROWS / (16 * WAVES), WAVES * 32, 0, stream>>>(
      opos, orel, ohs, h0, Wemb, bemb, Wih, bih, bhh, Wpred,
      ws_img, ws_whh, out);
}
